// GeoOperatorLinear_37048387895900
// MI455X (gfx1250) — matrix operand order repaired and checked by reading, not compiled, not run on hardware
//
#include <hip/hip_runtime.h>
#include <math.h>

// ---------- vector types for WMMA ----------
typedef __attribute__((ext_vector_type(16))) __bf16     v16bf;
typedef __attribute__((ext_vector_type(8)))  float      v8f;
typedef __attribute__((ext_vector_type(8)))  unsigned   v8u;

static __device__ inline unsigned f2bf1(float f) {
    unsigned u = __float_as_uint(f);
    return (u + 0x7FFFu + ((u >> 16) & 1u)) >> 16;   // RNE bf16
}
static __device__ inline unsigned pack2bf(float lo, float hi) {
    return f2bf1(lo) | (f2bf1(hi) << 16);
}
static __device__ inline v16bf asbf(v8u u) { return __builtin_bit_cast(v16bf, u); }

static __device__ inline v8f wmma_bf16(v16bf a, v16bf b, v8f c) {
    return __builtin_amdgcn_wmma_f32_16x16x32_bf16(false, a, false, b, (short)0, c, false, false);
}

// A-layout slot for bf16 16x32 fragments staged in LDS:
// lane = (row%16) + 16*((k%16)>=8), dword v packs K=2v.. pair, v+=4 for k%32>=16
static __device__ inline int a_lane(int row, int koff) {
    return (row & 15) + (((koff & 15) >= 8) ? 16 : 0);
}
static __device__ inline int a_v(int koff) {
    return ((koff & 7) >> 1) + ((koff & 16) ? 4 : 0);
}

// Build a WMMA A fragment from a 16-float row vector (K=16 real, K=16..31 zero)
static __device__ inline v16bf frag_from_rowvec(const float* rowv, int lane) {
    int base = (lane >> 4) ? 8 : 0;     // lanes 16-31 hold K=8..15
    v8u u;
    u[0] = pack2bf(rowv[base + 0], rowv[base + 1]);
    u[1] = pack2bf(rowv[base + 2], rowv[base + 3]);
    u[2] = pack2bf(rowv[base + 4], rowv[base + 5]);
    u[3] = pack2bf(rowv[base + 6], rowv[base + 7]);
    u[4] = 0u; u[5] = 0u; u[6] = 0u; u[7] = 0u;
    return asbf(u);
}

#define D_DIM 1024
#define S_DIM 4096
#define B_DIM 4

// ================= prep kernels: pack operands into exact WMMA B layouts =================

// W_base [1024,1024] -> WB[kt(32)][nt(64)][lane(32)][v(8)] bf16 dwords (2MB)
__global__ void pack_wb(const float* __restrict__ W, unsigned* __restrict__ WB) {
    int idx = blockIdx.x * 256 + threadIdx.x;          // < 524288
    int v = idx & 7, lane = (idx >> 3) & 31, nt = (idx >> 8) & 63, kt = idx >> 14;
    int n = nt * 16 + (lane & 15);
    int k = kt * 32 + ((lane >> 4) ? 8 : 0) + 2 * (v & 3) + ((v & 4) ? 16 : 0);
    WB[idx] = pack2bf(W[(size_t)k * D_DIM + n], W[(size_t)(k + 1) * D_DIM + n]);
}

// Two [8,1024] matrices as 16 columns, K=1024: dst[kt(32)][lane(32)][v(8)]  (32KB)
__global__ void pack_kb(const float* __restrict__ M1, const float* __restrict__ M2,
                        unsigned* __restrict__ dst) {
    int idx = blockIdx.x * 256 + threadIdx.x;          // < 8192
    int v = idx & 7, lane = (idx >> 3) & 31, kt = idx >> 8;
    int j = lane & 15;
    int k = kt * 32 + ((lane >> 4) ? 8 : 0) + 2 * (v & 3) + ((v & 4) ? 16 : 0);
    const float* M = (j < 8) ? (M1 + j * D_DIM) : (M2 + (j - 8) * D_DIM);
    dst[idx] = pack2bf(M[k], M[k + 1]);
}

// Two [8,1024] matrices stacked as 16 K-rows (zero-padded to 32), N=1024:
// dst[nt(64)][lane(32)][v(8)]  (64KB)
__global__ void pack_nb(const float* __restrict__ M1, const float* __restrict__ M2,
                        unsigned* __restrict__ dst) {
    int idx = blockIdx.x * 256 + threadIdx.x;          // < 16384
    int v = idx & 7, lane = (idx >> 3) & 31, nt = idx >> 8;
    if (v >= 4) { dst[idx] = 0u; return; }             // K=16..31 are zero pad
    int col = nt * 16 + (lane & 15);
    int k0 = 8 * (lane >> 4) + 2 * v, k1 = k0 + 1;
    float f0 = (k0 < 8) ? M1[k0 * D_DIM + col] : M2[(k0 - 8) * D_DIM + col];
    float f1 = (k1 < 8) ? M1[k1 * D_DIM + col] : M2[(k1 - 8) * D_DIM + col];
    dst[idx] = pack2bf(f0, f1);
}

// ================= pooling =================
__global__ void pool_partial(const float* __restrict__ x, float* __restrict__ partial) {
    int bd = (blockIdx.x & 15) * 256 + threadIdx.x;    // 0..4095
    int chunk = blockIdx.x >> 4;                       // 0..31
    int b = bd >> 10, d = bd & 1023;
    const float* p = x + ((size_t)b * S_DIM + chunk * 128) * D_DIM + d;
    float s = 0.f;
    for (int i = 0; i < 128; ++i) s += p[(size_t)i * D_DIM];
    partial[chunk * 4096 + bd] = s;
}

__global__ void pool_reduce(const float* __restrict__ partial, float* __restrict__ pooled) {
    int bd = blockIdx.x * 256 + threadIdx.x;
    float s = 0.f;
    for (int c = 0; c < 32; ++c) s += partial[c * 4096 + bd];
    pooled[bd] = s * (1.f / (float)S_DIM);
}

// ================= conditioner: LN -> MLP(GELU) -> gate heads =================
__global__ __launch_bounds__(128) void conditioner(
    const float* __restrict__ pooled, const float* __restrict__ ln_g, const float* __restrict__ ln_b,
    const float* __restrict__ Wc1, const float* __restrict__ bc1,
    const float* __restrict__ Wc2, const float* __restrict__ bc2,
    const float* __restrict__ Wrg, const float* __restrict__ brg,
    const float* __restrict__ Wrc, const float* __restrict__ brc,
    const float* __restrict__ Wsg, const float* __restrict__ bsg,
    const float* __restrict__ Wlc, const float* __restrict__ blc,
    float* __restrict__ gates) {
    __shared__ float red[128];
    __shared__ float z[1024];
    __shared__ float h1[128];
    __shared__ float h[128];
    int t = threadIdx.x;
    for (int b = 0; b < B_DIM; ++b) {
        const float* pb = pooled + b * D_DIM;
        float s = 0.f, s2 = 0.f;
        for (int i = t; i < D_DIM; i += 128) { float v = pb[i]; s += v; s2 += v * v; }
        red[t] = s; __syncthreads();
        for (int o = 64; o > 0; o >>= 1) { if (t < o) red[t] += red[t + o]; __syncthreads(); }
        float mean = red[0] * (1.f / (float)D_DIM); __syncthreads();
        red[t] = s2; __syncthreads();
        for (int o = 64; o > 0; o >>= 1) { if (t < o) red[t] += red[t + o]; __syncthreads(); }
        float var = red[0] * (1.f / (float)D_DIM) - mean * mean; __syncthreads();
        float inv = rsqrtf(var + 1e-5f);
        for (int i = t; i < D_DIM; i += 128) z[i] = (pb[i] - mean) * inv * ln_g[i] + ln_b[i];
        __syncthreads();
        float a = bc1[t];
        for (int i = 0; i < D_DIM; ++i) a += z[i] * Wc1[i * 128 + t];
        float a3 = a * a * a;  // tanh-approx GELU (jax default)
        h1[t] = 0.5f * a * (1.f + tanhf(0.7978845608028654f * (a + 0.044715f * a3)));
        __syncthreads();
        float hh = bc2[t];
        for (int i = 0; i < 128; ++i) hh += h1[i] * Wc2[i * 128 + t];
        h[t] = hh; __syncthreads();
        if (t < 25) {
            float acc;
            if (t == 0) {
                acc = brg[0];
                for (int i = 0; i < 128; ++i) acc += h[i] * Wrg[i];
                acc = 1.f / (1.f + __expf(-acc));          // sigmoid rot_gate
            } else if (t < 9) {
                int j = t - 1; acc = brc[j];
                for (int i = 0; i < 128; ++i) acc += h[i] * Wrc[i * 8 + j];
                acc = tanhf(acc);
            } else if (t < 17) {
                int j = t - 9; acc = bsg[j];
                for (int i = 0; i < 128; ++i) acc += h[i] * Wsg[i * 8 + j];
                acc = tanhf(acc);
            } else {
                int j = t - 17; acc = blc[j];
                for (int i = 0; i < 128; ++i) acc += h[i] * Wlc[i * 8 + j];
                acc = tanhf(acc);
            }
            gates[b * 25 + t] = acc;
        }
        __syncthreads();
    }
}

// ================= fused main kernel =================
// Per block: 64 rows of (B*S) x D. 256 threads = 8 waves.
// LDS: Abuf bf16 A-layout [4 strip][32 kt][32 lane][8 dw] (128KB) + rowc/rowc2 (4KB each)
__global__ __launch_bounds__(256) void fused_main(
    const float* __restrict__ x, const float* __restrict__ b_base,
    const unsigned* __restrict__ WB, const unsigned* __restrict__ BorthB,
    const unsigned* __restrict__ UBm, const unsigned* __restrict__ CTB,
    const unsigned* __restrict__ VB, const float* __restrict__ gates,
    float* __restrict__ out) {
    extern __shared__ char smem[];
    unsigned* Abuf  = (unsigned*)smem;                    // 32768 dwords
    float*    rowc  = (float*)(smem + 131072);            // [64][16]
    float*    rowc2 = (float*)(smem + 131072 + 4096);     // [64][16]

    const int tid = threadIdx.x;
    const int lane = tid & 31;
    const int wave = tid >> 5;
    const int rowBase = blockIdx.x * 64;
    const int batch = rowBase >> 12;                      // / S_DIM
    const float* gb = gates + batch * 25;

    // ---- P0: stage x tile -> bf16 A-layout in LDS
    {
        int r = tid >> 2;                                 // 0..63
        int q = tid & 3;
        int strip = r >> 4;
        const float* xrow = x + (size_t)(rowBase + r) * D_DIM + q * 256;
        for (int p = 0; p < 128; ++p) {
            int d = q * 256 + p * 2;
            float2 v = *(const float2*)(xrow + p * 2);
            int kt = d >> 5, koff = d & 31;
            Abuf[(((strip << 5) + kt) * 32 + a_lane(r, koff)) * 8 + a_v(koff)] = pack2bf(v.x, v.y);
        }
    }
    __syncthreads();

    // ---- P1: pa/pb = x @ [A_orth;B_orth]^T via WMMA; build skew rot coefficients
    if (wave < 4) {
        int strip = wave;
        v8f c = {};
        for (int kt = 0; kt < 32; ++kt) {
            v8u au = *(const v8u*)&Abuf[((strip * 32 + kt) * 32 + lane) * 8];
            v8u bu = *(const v8u*)&BorthB[(kt * 32 + lane) * 8];
            c = wmma_bf16(asbf(au), asbf(bu), c);
        }
        float g = gb[0];                                  // rot_gate (ROT_SCALE=1)
        int colj = lane & 15, half = lane >> 4;
        for (int i = 0; i < 8; ++i) {
            float ci = c[i];
            float partner = __shfl_xor(ci, 8, 32);        // pa_j <-> pb_j across cols
            float val = (colj < 8) ? (g * gb[1 + colj] * partner)        //  g*c_j*pb_j
                                   : (-g * gb[1 + colj - 8] * partner);  // -g*c_j*pa_j
            int row = strip * 16 + i + 8 * half;
            rowc[row * 16 + colj] = val;
        }
    }
    __syncthreads();

    // ---- P2: rot = rowvec @ [A_orth;B_orth]; restage x_rot = x + rot as bf16
    {
        int strip = wave & 3;
        int nt0 = (wave >> 2) * 32;
        int colj = lane & 15, half = lane >> 4;
        v16bf afrag = frag_from_rowvec(&rowc[(strip * 16 + colj) * 16], lane);
        unsigned short* Ah = (unsigned short*)smem;       // halfword view of Abuf
        for (int ntl = 0; ntl < 32; ++ntl) {
            int nt = nt0 + ntl;
            v8u bu = *(const v8u*)&CTB[(nt * 32 + lane) * 8];
            v8f c = {};
            c = wmma_bf16(afrag, asbf(bu), c);
            int col = nt * 16 + colj;
            int kt = col >> 5, koff = col & 31;
            int vv = a_v(koff);
            for (int i = 0; i < 8; ++i) {
                int rloc = strip * 16 + i + 8 * half;
                float xr = x[(size_t)(rowBase + rloc) * D_DIM + col] + c[i];
                int slot = (((strip << 5) + kt) * 32 + a_lane(rloc, koff)) * 8 + vv;
                Ah[slot * 2 + (koff & 1)] = (unsigned short)f2bf1(xr);
            }
        }
    }
    __syncthreads();

    // ---- P3: s,t = x_rot @ [U_spec;U_lr]^T via WMMA; gated residual weights
    if (wave < 4) {
        int strip = wave;
        v8f c = {};
        for (int kt = 0; kt < 32; ++kt) {
            v8u au = *(const v8u*)&Abuf[((strip * 32 + kt) * 32 + lane) * 8];
            v8u bu = *(const v8u*)&UBm[(kt * 32 + lane) * 8];
            c = wmma_bf16(asbf(au), asbf(bu), c);
        }
        int colj = lane & 15, half = lane >> 4;
        for (int i = 0; i < 8; ++i) {
            float ci = c[i];
            float val = (colj < 8) ? (gb[9 + colj] * ci)         // spec_gate * s  (scales=1)
                                   : (gb[17 + colj - 8] * ci);   // lr_coeff  * t
            int row = strip * 16 + i + 8 * half;
            rowc2[row * 16 + colj] = val;
        }
    }
    __syncthreads();

    // ---- P4: main GEMM  out = x_rot @ W_base  (+ V residual as extra K-chunk) + b_base
    {
        int strip = wave & 3;
        int colj = lane & 15, half = lane >> 4;
        v16bf a2 = frag_from_rowvec(&rowc2[(strip * 16 + colj) * 16], lane);
        for (int ng = 0; ng < 4; ++ng) {
            int ntbase = (wave >> 2) * 32 + ng * 8;
            v8f acc[8];
#pragma unroll
            for (int j = 0; j < 8; ++j) acc[j] = (v8f){};
            for (int kt = 0; kt < 32; ++kt) {
                v8u au = *(const v8u*)&Abuf[((strip * 32 + kt) * 32 + lane) * 8];
                v16bf a = asbf(au);
#pragma unroll
                for (int j = 0; j < 8; ++j) {
                    v8u bu = *(const v8u*)&WB[(((kt << 6) + ntbase + j) * 32 + lane) * 8];
                    acc[j] = wmma_bf16(a, asbf(bu), acc[j]);
                }
            }
#pragma unroll
            for (int j = 0; j < 8; ++j) {                 // V_spec/V_lr residual chunk
                v8u bu = *(const v8u*)&VB[((ntbase + j) * 32 + lane) * 8];
                acc[j] = wmma_bf16(a2, asbf(bu), acc[j]);
            }
#pragma unroll
            for (int j = 0; j < 8; ++j) {
                int col = (ntbase + j) * 16 + colj;
                float bb = b_base[col];
#pragma unroll
                for (int i = 0; i < 8; ++i) {
                    int rloc = strip * 16 + i + 8 * half;
                    out[(size_t)(rowBase + rloc) * D_DIM + col] = acc[j][i] + bb;
                }
            }
        }
    }
}

// ================= host launcher =================
extern "C" void kernel_launch(void* const* d_in, const int* in_sizes, int n_in,
                              void* d_out, int out_size, void* d_ws, size_t ws_size,
                              hipStream_t stream) {
    (void)in_sizes; (void)n_in; (void)out_size; (void)ws_size;
    const float* x      = (const float*)d_in[0];
    const float* W_base = (const float*)d_in[1];
    const float* b_base = (const float*)d_in[2];
    const float* ln_g   = (const float*)d_in[3];
    const float* ln_b   = (const float*)d_in[4];
    const float* Wc1    = (const float*)d_in[5];
    const float* bc1    = (const float*)d_in[6];
    const float* Wc2    = (const float*)d_in[7];
    const float* bc2    = (const float*)d_in[8];
    const float* Wrg    = (const float*)d_in[9];
    const float* brg    = (const float*)d_in[10];
    const float* Wrc    = (const float*)d_in[11];
    const float* brc    = (const float*)d_in[12];
    const float* Wsg    = (const float*)d_in[13];
    const float* bsg    = (const float*)d_in[14];
    const float* Wlc    = (const float*)d_in[15];
    const float* blc    = (const float*)d_in[16];
    const float* U_spec = (const float*)d_in[17];
    const float* V_spec = (const float*)d_in[18];
    const float* U_lr   = (const float*)d_in[19];
    const float* V_lr   = (const float*)d_in[20];
    const float* A_orth = (const float*)d_in[21];
    const float* B_orth = (const float*)d_in[22];
    float* out = (float*)d_out;

    char* ws = (char*)d_ws;
    unsigned* WB      = (unsigned*)(ws + 0);          // 2 MB
    unsigned* BorthB  = (unsigned*)(ws + 2097152);    // 32 KB
    unsigned* UBm     = (unsigned*)(ws + 2129920);    // 32 KB
    unsigned* CTB     = (unsigned*)(ws + 2162688);    // 64 KB
    unsigned* VB      = (unsigned*)(ws + 2228224);    // 64 KB
    float*    partial = (float*)(ws + 2293760);       // 512 KB
    float*    pooled  = (float*)(ws + 2818048);       // 16 KB
    float*    gates   = (float*)(ws + 2834432);       // 100 floats

    pack_wb<<<2048, 256, 0, stream>>>(W_base, WB);
    pack_kb<<<32, 256, 0, stream>>>(A_orth, B_orth, BorthB);
    pack_kb<<<32, 256, 0, stream>>>(U_spec, U_lr, UBm);
    pack_nb<<<64, 256, 0, stream>>>(A_orth, B_orth, CTB);
    pack_nb<<<64, 256, 0, stream>>>(V_spec, V_lr, VB);

    pool_partial<<<512, 256, 0, stream>>>(x, partial);
    pool_reduce<<<16, 256, 0, stream>>>(partial, pooled);
    conditioner<<<1, 128, 0, stream>>>(pooled, ln_g, ln_b, Wc1, bc1, Wc2, bc2,
                                       Wrg, brg, Wrc, brc, Wsg, bsg, Wlc, blc, gates);

    size_t shmem = 131072 + 4096 + 4096;              // 136 KB dynamic LDS
    hipFuncSetAttribute(reinterpret_cast<const void*>(fused_main),
                        hipFuncAttributeMaxDynamicSharedMemorySize, (int)shmem);
    fused_main<<<256, 256, shmem, stream>>>(x, b_base, WB, BorthB, UBm, CTB, VB, gates, out);
}
